// My_GATConv_3607772529308
// MI455X (gfx1250) — matrix operand order repaired and checked by reading, not compiled, not run on hardware
//
#include <hip/hip_runtime.h>
#include <hip/hip_bf16.h>

#define NN   50000
#define EE   500000
#define ET   (EE + NN)     // edges + self loops
#define DD   64
#define HH   8
#define COLS 512           // H*D
#define NEG_SLOPE 0.2f
#define EPSN 1e-5f

typedef __attribute__((ext_vector_type(16))) __bf16 v16bf;
typedef __attribute__((ext_vector_type(8)))  __bf16 v8bf;
typedef __attribute__((ext_vector_type(8)))  float  v8f;
typedef __attribute__((ext_vector_type(4)))  unsigned int u32x4;
typedef __attribute__((ext_vector_type(8)))  int i32x8;
typedef __attribute__((ext_vector_type(4)))  int i32x4;

// ---------- conversions ----------
__global__ __launch_bounds__(256) void k_convx(const float* __restrict__ x,
                                               __bf16* __restrict__ xbf) {
    int i = blockIdx.x * 256 + threadIdx.x;
    if (i < NN * DD) xbf[i] = (__bf16)x[i];
}

// Wt[col*64 + k] = W[k*512 + col]  (bf16, transposed so B-frag K is contiguous)
__global__ __launch_bounds__(256) void k_convw(const float* __restrict__ W,
                                               __bf16* __restrict__ wt) {
    int i = blockIdx.x * 256 + threadIdx.x;
    if (i < DD * COLS) {
        int col = i >> 6, k = i & 63;
        wt[i] = (__bf16)W[(size_t)k * COLS + col];
    }
}

// ---------- per-layer init (deterministic: runs every call) ----------
__global__ __launch_bounds__(256) void k_init(float* __restrict__ m, float* __restrict__ z,
                                              float* __restrict__ out, float* __restrict__ scal) {
    int i = blockIdx.x * 256 + threadIdx.x;
    if (i < NN * HH) { m[i] = -3.0e38f; z[i] = 0.f; }
    if (i < NN * DD) out[i] = 0.f;
    if (i < 4)       scal[i] = 0.f;
}

// ---------- WMMA GEMM: xw[N,512] = xbf[N,64] @ W[64,512] ----------
// block = 4 waves; 16-row tile per block, 128 output cols per wave.
// B operand (Wt, 64 KB bf16) is DMA'd into LDS once per block by the TDM.
__global__ __launch_bounds__(128) void k_gemm(const __bf16* __restrict__ xbf,
                                              const __bf16* __restrict__ wt,   // [512][64]
                                              float* __restrict__ xw) {
    __shared__ __attribute__((aligned(16))) __bf16 lds_w[DD * COLS];   // 64 KB

    const int wave = threadIdx.x >> 5;
    const int lane = threadIdx.x & 31;
    const int row0 = blockIdx.x * 16;
    const int half = lane >> 4;      // 0: lanes 0-15, 1: lanes 16-31
    const int lm   = lane & 15;

    if (wave == 0) {
        // Tensor DMA descriptor (CDNA5 ISA): 2-D tensor 64x512 bf16,
        // tile = whole tensor, contiguous -> linear 64 KB copy global->LDS.
        unsigned lds_off = (unsigned)(size_t)&lds_w[0];   // flat LDS addr: low 32 bits = LDS offset
        unsigned long long ga = (unsigned long long)(size_t)wt;
        u32x4 g0 = { 1u,                                          // count=1, user mode
                     lds_off,                                     // lds_addr (bytes)
                     (unsigned)(ga & 0xFFFFFFFFull),              // global_addr[31:0]
                     (unsigned)((ga >> 32) & 0x1FFFFFFull) | (2u << 30) }; // addr[56:32] | type=2
        i32x8 g1 = { 0x00010000,            // workgroup_mask=0, data_size=1 (2 bytes)
                     (int)(64u  << 16),     // tensor_dim0 = 64 (low16 in bits 63:48)
                     (int)(512u << 16),     // tensor_dim1 = 512 (low16 in bits 95:80)
                     (int)(64u  << 16),     // tile_dim0 = 64 (bits 127:112)
                     512,                   // tile_dim1 = 512 (bits 143:128)
                     64,                    // tensor_dim0_stride = 64 (bits 191:160)
                     0, 0 };
        i32x4 gz4 = { 0, 0, 0, 0 };
        i32x8 gz8 = { 0, 0, 0, 0, 0, 0, 0, 0 };
        __builtin_amdgcn_tensor_load_to_lds(g0, g1, gz4, gz4, gz8, 0);
        __builtin_amdgcn_s_wait_tensorcnt(0);
    }
    __syncthreads();

    v8f acc[8];
#pragma unroll
    for (int t = 0; t < 8; ++t) acc[t] = (v8f){0.f,0.f,0.f,0.f,0.f,0.f,0.f,0.f};

#pragma unroll
    for (int ks = 0; ks < 2; ++ks) {
        // A fragment (16x32 bf16): lane holds row (row0+lm), K = kb..kb+7 and kb+16..kb+23
        const int kbA = half * 8 + ks * 32;
        const __bf16* ap = xbf + (size_t)(row0 + lm) * DD + kbA;
        v8bf alo = *(const v8bf*)(ap);
        v8bf ahi = *(const v8bf*)(ap + 16);
        v16bf a;
#pragma unroll
        for (int e = 0; e < 8; ++e) { a[e] = alo[e]; a[8 + e] = ahi[e]; }

#pragma unroll
        for (int t = 0; t < 8; ++t) {
            const int col = wave * 128 + t * 16 + lm;
            const int kbB = half * 8 + ks * 32;
            const __bf16* bp = lds_w + (size_t)col * DD + kbB;   // ds_load_b128 x2
            v8bf blo = *(const v8bf*)(bp);
            v8bf bhi = *(const v8bf*)(bp + 16);
            v16bf b;
#pragma unroll
            for (int e = 0; e < 8; ++e) { b[e] = blo[e]; b[8 + e] = bhi[e]; }
            acc[t] = __builtin_amdgcn_wmma_f32_16x16x32_bf16(
                false, a, false, b, (short)0, acc[t], false, false);
        }
    }
    // D layout: VGPR j -> row = j + half*8, col = lane&15 within tile
#pragma unroll
    for (int t = 0; t < 8; ++t) {
        const int col = wave * 128 + t * 16 + lm;
#pragma unroll
        for (int j = 0; j < 8; ++j) {
            const int row = row0 + j + half * 8;
            xw[(size_t)row * COLS + col] = acc[t][j];
        }
    }
}

// ---------- attention dots: a_src/a_dst [N,H] ----------
__global__ __launch_bounds__(256) void k_att(const float* __restrict__ xw,
                                             const float* __restrict__ att_s,
                                             const float* __restrict__ att_d,
                                             float* __restrict__ a_src,
                                             float* __restrict__ a_dst) {
    const int wave = threadIdx.x >> 5, lane = threadIdx.x & 31;
    const int n = blockIdx.x * 8 + wave;
    if (n >= NN) return;
    const float* row = xw + (size_t)n * COLS;
    float ps[HH], pd[HH];
#pragma unroll
    for (int h = 0; h < HH; ++h) {
        int c0 = h * 64 + lane, c1 = c0 + 32;
        float v0 = row[c0], v1 = row[c1];
        ps[h] = v0 * att_s[c0] + v1 * att_s[c1];
        pd[h] = v0 * att_d[c0] + v1 * att_d[c1];
    }
#pragma unroll
    for (int off = 16; off > 0; off >>= 1) {
#pragma unroll
        for (int h = 0; h < HH; ++h) {
            ps[h] += __shfl_xor(ps[h], off, 32);
            pd[h] += __shfl_xor(pd[h], off, 32);
        }
    }
    if (lane == 0) {
#pragma unroll
        for (int h = 0; h < HH; ++h) { a_src[n * HH + h] = ps[h]; a_dst[n * HH + h] = pd[h]; }
    }
}

// float atomic max via monotone integer mapping (init = -3e38)
__device__ inline void atomicMaxF(float* addr, float val) {
    if (val >= 0.f) atomicMax((int*)addr, __float_as_int(val));
    else            atomicMin((unsigned int*)addr, __float_as_uint(val));
}

// ---------- edge logits + segment max ----------
__global__ __launch_bounds__(256) void k_logit(const int* __restrict__ ei,
                                               const float* __restrict__ a_src,
                                               const float* __restrict__ a_dst,
                                               float* __restrict__ ebuf,
                                               float* __restrict__ m) {
    int e = blockIdx.x * 256 + threadIdx.x;
    if (e >= ET) return;
    int s, d;
    if (e < EE) { s = ei[e]; d = ei[EE + e]; } else { s = d = e - EE; }
#pragma unroll
    for (int h = 0; h < HH; ++h) {
        float v = a_src[s * HH + h] + a_dst[d * HH + h];
        v = v > 0.f ? v : NEG_SLOPE * v;
        ebuf[(size_t)e * HH + h] = v;
        atomicMaxF(&m[d * HH + h], v);
    }
}

// ---------- p = exp(e - m[dst]); z[dst] += p ----------
__global__ __launch_bounds__(256) void k_soft(const int* __restrict__ ei,
                                              const float* __restrict__ m,
                                              float* __restrict__ pbuf,
                                              float* __restrict__ z) {
    int idx = blockIdx.x * 256 + threadIdx.x;
    if (idx >= ET * HH) return;
    int e = idx >> 3, h = idx & 7;
    int d = (e < EE) ? ei[EE + e] : e - EE;
    float p = __expf(pbuf[idx] - m[d * HH + h]);
    pbuf[idx] = p;
    atomicAdd(&z[d * HH + h], p);
}

// ---------- aggregate (fused head-mean): out[dst,c] += (1/8) sum_h alpha_h*xw[src,h*64+c]
__global__ __launch_bounds__(256) void k_agg(const int* __restrict__ ei,
                                             const float* __restrict__ pbuf,
                                             const float* __restrict__ z,
                                             const float* __restrict__ xw,
                                             float* __restrict__ out) {
    const int wave = threadIdx.x >> 5, lane = threadIdx.x & 31;
    const int e = blockIdx.x * 8 + wave;
    if (e >= ET) return;
    int s, d;
    if (e < EE) { s = ei[e]; d = ei[EE + e]; } else { s = d = e - EE; }
    const float* row = xw + (size_t)s * COLS;
    __builtin_prefetch(row, 0, 1);   // global_prefetch of the L2-resident source row
    float aval = 0.f;
    if (lane < HH) aval = pbuf[(size_t)e * HH + lane] / z[d * HH + lane];
    float acc0 = 0.f, acc1 = 0.f;
#pragma unroll
    for (int h = 0; h < HH; ++h) {
        float al = __shfl(aval, h, 32);
        acc0 += al * row[h * 64 + lane];
        acc1 += al * row[h * 64 + lane + 32];
    }
    atomicAdd(&out[(size_t)d * DD + lane],       0.125f * acc0);
    atomicAdd(&out[(size_t)d * DD + lane + 32],  0.125f * acc1);
}

// ---------- residual + bias; accumulate global sum / sumsq ----------
__global__ __launch_bounds__(256) void k_res(const float* __restrict__ xin,
                                             const float* __restrict__ out,
                                             const float* __restrict__ bias,
                                             float* __restrict__ hbuf,
                                             float* __restrict__ scal) {
    int i = blockIdx.x * 256 + threadIdx.x;   // i < N*D exactly
    float v = xin[i] + out[i] + bias[i & 63];
    hbuf[i] = v;
    float s = v, s2 = v * v;
#pragma unroll
    for (int off = 16; off > 0; off >>= 1) {
        s  += __shfl_xor(s,  off, 32);
        s2 += __shfl_xor(s2, off, 32);
    }
    if ((threadIdx.x & 31) == 0) { atomicAdd(&scal[0], s); atomicAdd(&scal[1], s2); }
}

// ---------- graph-norm + relu; emit f32 (+ bf16 for next GEMM) ----------
__global__ __launch_bounds__(256) void k_norm(const float* __restrict__ hbuf,
                                              const float* __restrict__ g,
                                              const float* __restrict__ b,
                                              const float* __restrict__ scal,
                                              float* __restrict__ hout,
                                              __bf16* __restrict__ hbf) {
    int i = blockIdx.x * 256 + threadIdx.x;
    if (i >= NN * DD) return;
    const float inv = 1.0f / (float)(NN * DD);
    float mu  = scal[0] * inv;
    float var = scal[1] * inv - mu * mu;
    float r   = rsqrtf(var + EPSN);
    float v   = (hbuf[i] - mu) * r * g[i & 63] + b[i & 63];
    v = fmaxf(v, 0.f);
    hout[i] = v;
    hbf[i]  = (__bf16)v;
}

// ---------- host ----------
static inline char* wsalloc(char*& p, size_t bytes) {
    char* r = p;
    p += (bytes + 255) & ~(size_t)255;
    return r;
}

extern "C" void kernel_launch(void* const* d_in, const int* in_sizes, int n_in,
                              void* d_out, int out_size, void* d_ws, size_t ws_size,
                              hipStream_t stream) {
    (void)in_sizes; (void)n_in; (void)out_size; (void)ws_size;
    const float* x       = (const float*)d_in[0];
    const int*   ei      = (const int*)  d_in[1];
    const float* W1      = (const float*)d_in[2];
    const float* att_s1  = (const float*)d_in[3];
    const float* att_d1  = (const float*)d_in[4];
    const float* bias1   = (const float*)d_in[5];
    const float* g1      = (const float*)d_in[6];
    const float* b1      = (const float*)d_in[7];
    const float* W2      = (const float*)d_in[8];
    const float* att_s2  = (const float*)d_in[9];
    const float* att_d2  = (const float*)d_in[10];
    const float* bias2   = (const float*)d_in[11];
    const float* g2      = (const float*)d_in[12];
    const float* b2      = (const float*)d_in[13];
    float* outp = (float*)d_out;

    char* p = (char*)d_ws;
    __bf16* xbf   = (__bf16*)wsalloc(p, (size_t)NN * DD * 2);
    __bf16* hbf   = (__bf16*)wsalloc(p, (size_t)NN * DD * 2);
    __bf16* wt1   = (__bf16*)wsalloc(p, (size_t)DD * COLS * 2);
    __bf16* wt2   = (__bf16*)wsalloc(p, (size_t)DD * COLS * 2);
    float*  xw    = (float*) wsalloc(p, (size_t)NN * COLS * 4);
    float*  a_src = (float*) wsalloc(p, (size_t)NN * HH * 4);
    float*  a_dst = (float*) wsalloc(p, (size_t)NN * HH * 4);
    float*  m     = (float*) wsalloc(p, (size_t)NN * HH * 4);
    float*  z     = (float*) wsalloc(p, (size_t)NN * HH * 4);
    float*  pbuf  = (float*) wsalloc(p, (size_t)ET * HH * 4);
    float*  obuf  = (float*) wsalloc(p, (size_t)NN * DD * 4);
    float*  hbuf  = (float*) wsalloc(p, (size_t)NN * DD * 4);
    float*  scal  = (float*) wsalloc(p, 256);

    const int gND   = (NN * DD + 255) / 256;      // 12500
    const int gW    = (DD * COLS + 255) / 256;    // 128
    const int gE    = (ET + 255) / 256;
    const int gEH   = (ET * HH + 255) / 256;
    const int gAGG  = (ET + 7) / 8;
    const int gATT  = (NN + 7) / 8;
    const int gGEMM = NN / 16;                    // 3125

    k_convx<<<gND, 256, 0, stream>>>(x, xbf);
    k_convw<<<gW, 256, 0, stream>>>(W1, wt1);
    k_convw<<<gW, 256, 0, stream>>>(W2, wt2);

    // ---- layer 1 ----
    k_init <<<gND, 256, 0, stream>>>(m, z, obuf, scal);
    k_gemm <<<gGEMM, 128, 0, stream>>>(xbf, wt1, xw);
    k_att  <<<gATT, 256, 0, stream>>>(xw, att_s1, att_d1, a_src, a_dst);
    k_logit<<<gE, 256, 0, stream>>>(ei, a_src, a_dst, pbuf, m);
    k_soft <<<gEH, 256, 0, stream>>>(ei, m, pbuf, z);
    k_agg  <<<gAGG, 256, 0, stream>>>(ei, pbuf, z, xw, obuf);
    k_res  <<<gND, 256, 0, stream>>>(x, obuf, bias1, hbuf, scal);
    k_norm <<<gND, 256, 0, stream>>>(hbuf, g1, b1, scal, hbuf, hbf);

    // ---- layer 2 ----
    k_init <<<gND, 256, 0, stream>>>(m, z, obuf, scal);
    k_gemm <<<gGEMM, 128, 0, stream>>>(hbf, wt2, xw);
    k_att  <<<gATT, 256, 0, stream>>>(xw, att_s2, att_d2, a_src, a_dst);
    k_logit<<<gE, 256, 0, stream>>>(ei, a_src, a_dst, pbuf, m);
    k_soft <<<gEH, 256, 0, stream>>>(ei, m, pbuf, z);
    k_agg  <<<gAGG, 256, 0, stream>>>(ei, pbuf, z, xw, obuf);
    k_res  <<<gND, 256, 0, stream>>>(hbuf, obuf, bias2, hbuf, scal);
    k_norm <<<gND, 256, 0, stream>>>(hbuf, g2, b2, scal, outp, hbf);
}
